// MultiViewAgentCausalSlotPredictor_90744069030090
// MI455X (gfx1250) — compile-verified, matrix operand order checked by reading, not run on hardware
//
#include <hip/hip_runtime.h>
#include <hip/hip_bf16.h>
#include <stdint.h>

// ---------------------------------------------------------------------------
// MultiViewAgentCausalSlotPredictor — fused recurrent step kernel for gfx1250
//   Pass 1: pack_weights — f32 weights -> f16 WMMA B-fragments in d_ws (once
//           per launch; 3.33 MB, L2-resident for the whole main kernel).
//   Pass 2: slotpred_kernel — grid = B (one WG per batch element), 8 wave32.
//           Recurrent state (32x256 f32) lives in LDS across all TP=32 steps.
//           GEMMs: v_wmma_f32_16x16x32_f16; per wave: one N-column of tiles,
//           MTILES accumulators share each B-fragment (B loaded once).
// ---------------------------------------------------------------------------

typedef _Float16 f16;
typedef __attribute__((ext_vector_type(16))) _Float16 v16h;
typedef __attribute__((ext_vector_type(8)))  float    v8f;

#define NTHREADS 256
#define NWAVES   8

#define Bsz  256
#define TPn  32
#define Dn   256
#define Tn   32      // V*S tokens
#define Vn   4
#define Sn   8
#define An   32
#define MLPn 1024

// ---- weight-fragment arena in d_ws (units: fragments of 1 KB = 64 uint4) --
#define FR_WQP 0      // w_qp  N=256 K=512  -> 256 frags
#define FR_AWQ 256    // a w_in rows 0..255      128
#define FR_AWK 384
#define FR_AWV 512
#define FR_AWO 640
#define FR_AW1 768    // a w1 N=1024 K=256       512
#define FR_AW2 1280   // a w2 N=256  K=1024      512
#define FR_OWQ 1792
#define FR_OWK 1920
#define FR_OWV 2048
#define FR_OWO 2176
#define FR_OW1 2304
#define FR_OW2 2816
#define FR_TOTAL 3328 // 3,407,872 bytes of d_ws

// ---- LDS layout (bytes) ----------------------------------------------------
#define OFF_STATE 0         // 32x256 f32
#define OFF_CTX   32768     // 32x256 f32
#define OFF_XA    65536     // f16 GEMM-A staging (<=36x256)
#define OFF_KV16  83968     // f16 kv operand (<=36x256)
#define OFF_H     102400    // f16 FF hidden 28x1024
#define OFF_QH    159744    // f32 qh  <=28x256
#define OFF_KH    188416    // f32 kh / kv_obj pre-LN <=36x256
#define OFF_VH    225280    // f32 vh  <=36x256
#define OFF_Y     262144    // f32 GEMM out scratch <=28x256
#define OFF_AGN   290816    // f32 agent_next 4x256
#define OFF_OBJ   294912    // f32 obj_next 28x256
#define OFF_ACT   323584    // f32 act proj 256
#define SMEM_BYTES 324608

union H2 { f16 h[2]; uint32_t u; };

__device__ __forceinline__ float gelu_f(float x) {
    return 0.5f * x * (1.0f + erff(x * 0.70710678118654752f));
}

// ---------------------------------------------------------------------------
// Pack one weight matrix W[N,K] (f32) into WMMA B-fragment order (f16).
// Fragment (nt,kt) = 16x32 tile of B[k][n]=W[n][k]; 1 KB each:
//   bytes [0,512):  lane-major uint4 = B dwords q=0..3
//   bytes [512,1K): lane-major uint4 = B dwords q=4..7
// Lane l: n = nt*16 + (l&15); k0(q) = (q&3)*2 + (q>=4)*16 + (l>>4)*8.
// ---------------------------------------------------------------------------
__global__ void pack_weights(const float* __restrict__ W, int N, int K,
                             uint4* __restrict__ dst) {
    const int frag = blockIdx.x * NWAVES + (threadIdx.x >> 5);
    const int lane = threadIdx.x & 31;
    const int Ktiles = K >> 5;
    if (frag >= (N >> 4) * Ktiles) return;
    const int nt = frag / Ktiles, kt = frag - nt * Ktiles;
    const float* wp = W + (size_t)(nt * 16 + (lane & 15)) * K + kt * 32
                        + ((lane >> 4) << 3);
    uint32_t u[8];
    #pragma unroll
    for (int q = 0; q < 8; ++q) {
        const int k0 = ((q & 3) << 1) + ((q >> 2) << 4);
        float2 w2 = *(const float2*)(wp + k0);
        H2 h2; h2.h[0] = (f16)w2.x; h2.h[1] = (f16)w2.y;
        u[q] = h2.u;
    }
    uint4* out = dst + (size_t)frag * 64;
    out[lane]      = make_uint4(u[0], u[1], u[2], u[3]);
    out[lane + 32] = make_uint4(u[4], u[5], u[6], u[7]);
}

// ---------------------------------------------------------------------------
// y[M,N] = x[M,K](f16,LDS) @ W^T + bias, W pre-packed as B-fragments.
//   MTILES = ceil(Mt/16) (compile-time). Each wave owns whole N-columns:
//   one B-fragment pair feeds MTILES independent WMMAs.
//   mode 0: f32 out (stride N)     mode 2: gelu -> f16 out (stride N)
// ---------------------------------------------------------------------------
template <int MTILES>
__device__ void gemm(const f16* __restrict__ xA, int Mt, int K,
                     const uint4* __restrict__ fr0, const float* __restrict__ bias,
                     int N, int mode, float* __restrict__ yf, f16* __restrict__ yh) {
    const int wid  = threadIdx.x >> 5;
    const int lane = threadIdx.x & 31;
    const int half = lane >> 4;
    const int l16  = lane & 15;
    const int Ntiles = N >> 4;
    const int Ktiles = K >> 5;

    // Per-M-tile A row pointers (row clamped; OOB rows never stored).
    const f16* ap[MTILES];
    #pragma unroll
    for (int m = 0; m < MTILES; ++m) {
        const int arow = m * 16 + l16;
        ap[m] = xA + (size_t)(arow < Mt ? arow : 0) * K + (half << 3);
    }

    for (int nt = wid; nt < Ntiles; nt += NWAVES) {
        const uint4* fb = fr0 + (size_t)nt * Ktiles * 64 + lane;

        v8f acc[MTILES];
        #pragma unroll
        for (int m = 0; m < MTILES; ++m) acc[m] = (v8f){};

        for (int kb = 0; kb < K; kb += 32) {
            union { uint4 q[2]; v16h v; } Bv;
            Bv.q[0] = fb[0];
            Bv.q[1] = fb[32];
            if (kb + 32 < K) __builtin_prefetch(fb + 64, 0, 1);
            fb += 64;
            #pragma unroll
            for (int m = 0; m < MTILES; ++m) {
                union { uint4 q[2]; v16h v; } A;
                A.q[0] = *(const uint4*)(ap[m] + kb);        // k dwords 0..3
                A.q[1] = *(const uint4*)(ap[m] + kb + 16);   // k dwords 4..7
                acc[m] = __builtin_amdgcn_wmma_f32_16x16x32_f16(
                    false, A.v, false, Bv.v, (short)0, acc[m], false, false);
            }
        }

        const int n  = (nt << 4) + l16;
        const float bv = bias[n];
        #pragma unroll
        for (int m = 0; m < MTILES; ++m) {
            const int m0 = (m << 4) + (half << 3);
            #pragma unroll
            for (int r = 0; r < 8; ++r) {
                const int mm = m0 + r;
                if (mm < Mt) {
                    float val = acc[m][r] + bv;
                    if (mode == 2) { yh[(size_t)mm * N + n] = (f16)gelu_f(val); }
                    else           { yf[(size_t)mm * N + n] = val; }
                }
            }
        }
    }
}

// LayerNorm rows of x (f32, stride 256) -> f16 out (stride 256)
__device__ void layernorm_to_f16(const float* __restrict__ xin, int Mt,
                                 const float* __restrict__ g,
                                 const float* __restrict__ beta,
                                 f16* __restrict__ yout) {
    const int wid  = threadIdx.x >> 5;
    const int lane = threadIdx.x & 31;
    for (int m = wid; m < Mt; m += NWAVES) {
        const float* row = xin + (size_t)m * Dn;
        float s = 0.f, s2 = 0.f;
        #pragma unroll
        for (int i = 0; i < Dn / 32; ++i) {
            float v = row[lane + i * 32];
            s += v; s2 += v * v;
        }
        #pragma unroll
        for (int off = 16; off > 0; off >>= 1) {
            s  += __shfl_xor(s,  off, 32);
            s2 += __shfl_xor(s2, off, 32);
        }
        const float mean = s * (1.0f / Dn);
        const float var  = fmaxf(s2 * (1.0f / Dn) - mean * mean, 0.f);
        const float inv  = rsqrtf(var + 1e-5f);
        #pragma unroll
        for (int i = 0; i < Dn / 32; ++i) {
            const int c = lane + i * 32;
            yout[(size_t)m * Dn + c] = (f16)((row[c] - mean) * inv * g[c] + beta[c]);
        }
    }
}

// Streaming 2-pass softmax attention (H=8 heads, Dh=32). One thread per (q,h).
template <int MQ, int MK>
__device__ void attention(const float* __restrict__ qh, const float* __restrict__ kh,
                          const float* __restrict__ vh, f16* __restrict__ outp) {
    const int idx = threadIdx.x;
    if (idx >= MQ * 8) return;
    const int i = idx >> 3, h = idx & 7;
    const float* q = qh + (size_t)i * Dn + h * 32;
    float qreg[32];
    #pragma unroll
    for (int c = 0; c < 32; ++c) qreg[c] = q[c];

    const float scale = 0.17677669529663687f;  // 1/sqrt(32)
    float mx = -3.0e38f;
    #pragma unroll 1
    for (int j = 0; j < MK; ++j) {
        const float* kr = kh + (size_t)j * Dn + h * 32;
        float d = 0.f;
        #pragma unroll
        for (int c = 0; c < 32; ++c) d = fmaf(qreg[c], kr[c], d);
        mx = fmaxf(mx, d * scale);
    }
    float sum = 0.f;
    float o[32];
    #pragma unroll
    for (int c = 0; c < 32; ++c) o[c] = 0.f;
    #pragma unroll 1
    for (int j = 0; j < MK; ++j) {
        const float* kr = kh + (size_t)j * Dn + h * 32;
        const float* vr = vh + (size_t)j * Dn + h * 32;
        float d = 0.f;
        #pragma unroll
        for (int c = 0; c < 32; ++c) d = fmaf(qreg[c], kr[c], d);
        const float e = __expf(d * scale - mx);
        sum += e;
        #pragma unroll
        for (int c = 0; c < 32; ++c) o[c] = fmaf(e, vr[c], o[c]);
    }
    const float inv = 1.f / sum;
    #pragma unroll
    for (int c = 0; c < 32; ++c) outp[(size_t)i * Dn + h * 32 + c] = (f16)(o[c] * inv);
}

struct Params {
    const float *hist, *act;
    const float *w_act, *b_act, *b_qp;
    const float *a_qn_g, *a_qn_b, *a_kvn_g, *a_kvn_b, *a_ffn_g, *a_ffn_b;
    const float *a_b_in, *a_b_out, *a_b1, *a_b2;
    const float *o_qn_g, *o_qn_b, *o_kvn_g, *o_kvn_b, *o_ffn_g, *o_ffn_b;
    const float *o_b_in, *o_b_out, *o_b1, *o_b2;
    const float *view_emb, *slot_emb;
    const uint4 *frags;
    float *ps, *po, *pa;
};

__global__ __launch_bounds__(NTHREADS, 1)
void slotpred_kernel(Params p) {
    extern __shared__ char smem[];
    float* s_state = (float*)(smem + OFF_STATE);
    float* s_ctx   = (float*)(smem + OFF_CTX);
    f16*   s_xa    = (f16*)  (smem + OFF_XA);
    f16*   s_kv16  = (f16*)  (smem + OFF_KV16);
    f16*   s_h     = (f16*)  (smem + OFF_H);
    float* s_qh    = (float*)(smem + OFF_QH);
    float* s_kh    = (float*)(smem + OFF_KH);
    float* s_vh    = (float*)(smem + OFF_VH);
    float* s_y     = (float*)(smem + OFF_Y);
    float* s_agn   = (float*)(smem + OFF_AGN);
    float* s_obj   = (float*)(smem + OFF_OBJ);
    float* s_act   = (float*)(smem + OFF_ACT);

    const int tid = threadIdx.x;
    const int b   = blockIdx.x;
    const uint4* FR = p.frags;

    // state = history_slots[b, TH-1]   (b128 copy)
    const float* h0 = p.hist + ((size_t)b * 2 + 1) * (Tn * Dn);
    for (int i = tid * 4; i < Tn * Dn; i += NTHREADS * 4)
        *(float4*)(s_state + i) = *(const float4*)(h0 + i);
    __syncthreads();

    for (int t = 0; t < TPn; ++t) {
        // --- ctx = state + view_emb + slot_emb (vec4); act = action@w_act^T+b
        for (int i = tid * 4; i < Tn * Dn; i += NTHREADS * 4) {
            const int tok = i >> 8, d = i & 255;
            float4 st = *(const float4*)(s_state + i);
            float4 ve = *(const float4*)(p.view_emb + (tok >> 3) * Dn + d);
            float4 se = *(const float4*)(p.slot_emb + (tok & 7) * Dn + d);
            *(float4*)(s_ctx + i) = make_float4(st.x + ve.x + se.x, st.y + ve.y + se.y,
                                                st.z + ve.z + se.z, st.w + ve.w + se.w);
        }
        {
            const float* arow = p.act + ((size_t)b * TPn + t) * An;
            float s = p.b_act[tid];
            #pragma unroll
            for (int j = 0; j < An; ++j) s = fmaf(arow[j], p.w_act[tid * An + j], s);
            s_act[tid] = s;
        }
        __syncthreads();

        // --- kv_agent = LN(ctx);  q_in = [agent_ctx | act] (f16, stride 512)
        layernorm_to_f16(s_ctx, Tn, p.a_kvn_g, p.a_kvn_b, s_kv16);
        for (int i = tid; i < 4 * 512; i += NTHREADS) {
            const int r = i >> 9, c = i & 511;
            const float v = (c < Dn) ? s_ctx[(r * Sn + 7) * Dn + c] : s_act[c - Dn];
            s_xa[i] = (f16)v;
        }
        __syncthreads();

        // --- agent_query = q_in @ w_qp^T + b_qp ; q_a = LN(...)
        gemm<1>(s_xa, 4, 2 * Dn, FR + (size_t)FR_WQP * 64, p.b_qp, Dn, 0, s_y, nullptr);
        __syncthreads();
        layernorm_to_f16(s_y, 4, p.a_qn_g, p.a_qn_b, s_xa);
        __syncthreads();

        // --- agent MHA projections
        gemm<1>(s_xa,   4,  Dn, FR + (size_t)FR_AWQ * 64, p.a_b_in,          Dn, 0, s_qh, nullptr);
        gemm<2>(s_kv16, Tn, Dn, FR + (size_t)FR_AWK * 64, p.a_b_in + Dn,     Dn, 0, s_kh, nullptr);
        gemm<2>(s_kv16, Tn, Dn, FR + (size_t)FR_AWV * 64, p.a_b_in + 2 * Dn, Dn, 0, s_vh, nullptr);
        __syncthreads();
        attention<4, Tn>(s_qh, s_kh, s_vh, s_xa);
        __syncthreads();
        gemm<1>(s_xa, 4, Dn, FR + (size_t)FR_AWO * 64, p.a_b_out, Dn, 0, s_y, nullptr);
        __syncthreads();
        for (int i = tid; i < 4 * Dn; i += NTHREADS)
            s_agn[i] = s_state[((i >> 8) * Sn + 7) * Dn + (i & 255)] + s_y[i];
        __syncthreads();

        // --- agent FF
        layernorm_to_f16(s_agn, 4, p.a_ffn_g, p.a_ffn_b, s_xa);
        __syncthreads();
        gemm<1>(s_xa, 4, Dn, FR + (size_t)FR_AW1 * 64, p.a_b1, MLPn, 2, nullptr, s_h);
        __syncthreads();
        gemm<1>(s_h, 4, MLPn, FR + (size_t)FR_AW2 * 64, p.a_b2, Dn, 0, s_y, nullptr);
        __syncthreads();
        for (int i = tid; i < 4 * Dn; i += NTHREADS) s_agn[i] += s_y[i];
        __syncthreads();

        // --- kv_obj (pre-LN) into s_kh: [na_ctx(28) | ag_t+emb(4) | ag_n+emb(4)]
        for (int i = tid * 4; i < 36 * Dn; i += NTHREADS * 4) {
            const int r = i >> 8, d = i & 255;
            float4 v;
            if (r < 28) {
                const int vw = r / 7, s = r - vw * 7;
                v = *(const float4*)(s_ctx + (vw * Sn + s) * Dn + d);
            } else {
                const int vw = (r < 32) ? (r - 28) : (r - 32);
                const float* src = (r < 32) ? (s_state + (vw * Sn + 7) * Dn + d)
                                            : (s_agn + vw * Dn + d);
                float4 a  = *(const float4*)src;
                float4 ve = *(const float4*)(p.view_emb + vw * Dn + d);
                float4 se = *(const float4*)(p.slot_emb + 7 * Dn + d);
                v = make_float4(a.x + ve.x + se.x, a.y + ve.y + se.y,
                                a.z + ve.z + se.z, a.w + ve.w + se.w);
            }
            *(float4*)(s_kh + i) = v;
        }
        __syncthreads();
        layernorm_to_f16(s_kh, 36, p.o_kvn_g, p.o_kvn_b, s_kv16);   // kv_obj
        layernorm_to_f16(s_kh, 28, p.o_qn_g,  p.o_qn_b,  s_xa);     // q_o = LN(na_ctx)
        __syncthreads();

        // --- object MHA
        gemm<2>(s_xa,   28, Dn, FR + (size_t)FR_OWQ * 64, p.o_b_in,          Dn, 0, s_qh, nullptr);
        gemm<3>(s_kv16, 36, Dn, FR + (size_t)FR_OWK * 64, p.o_b_in + Dn,     Dn, 0, s_kh, nullptr);
        gemm<3>(s_kv16, 36, Dn, FR + (size_t)FR_OWV * 64, p.o_b_in + 2 * Dn, Dn, 0, s_vh, nullptr);
        __syncthreads();
        attention<28, 36>(s_qh, s_kh, s_vh, s_xa);
        __syncthreads();
        gemm<2>(s_xa, 28, Dn, FR + (size_t)FR_OWO * 64, p.o_b_out, Dn, 0, s_y, nullptr);
        __syncthreads();
        for (int i = tid; i < 28 * Dn; i += NTHREADS) {
            const int r = i >> 8, d = i & 255;
            const int vw = r / 7, s = r - vw * 7;
            s_obj[i] = s_state[(vw * Sn + s) * Dn + d] + s_y[i];
        }
        __syncthreads();

        // --- object FF
        layernorm_to_f16(s_obj, 28, p.o_ffn_g, p.o_ffn_b, s_xa);
        __syncthreads();
        gemm<2>(s_xa, 28, Dn, FR + (size_t)FR_OW1 * 64, p.o_b1, MLPn, 2, nullptr, s_h);
        __syncthreads();
        gemm<2>(s_h, 28, MLPn, FR + (size_t)FR_OW2 * 64, p.o_b2, Dn, 0, s_y, nullptr);
        __syncthreads();
        for (int i = tid; i < 28 * Dn; i += NTHREADS) s_obj[i] += s_y[i];
        __syncthreads();

        // --- outputs (b128 stores; this is the HBM roofline term) + state
        const size_t bt = (size_t)b * TPn + t;
        float* ps = p.ps + bt * (Tn * Dn);
        float* po = p.po + bt * (28 * Dn);
        float* pa = p.pa + bt * (4 * Dn);
        for (int i = tid * 4; i < Tn * Dn; i += NTHREADS * 4) {
            const int tok = i >> 8, d = i & 255;
            const int vw = tok >> 3, s = tok & 7;
            float4 v = (s == 7) ? *(const float4*)(s_agn + vw * Dn + d)
                                : *(const float4*)(s_obj + (vw * 7 + s) * Dn + d);
            *(float4*)(s_state + i) = v;
            *(float4*)(ps + i) = v;
        }
        for (int i = tid * 4; i < 28 * Dn; i += NTHREADS * 4)
            *(float4*)(po + i) = *(const float4*)(s_obj + i);
        for (int i = tid * 4; i < 4 * Dn; i += NTHREADS * 4)
            *(float4*)(pa + i) = *(const float4*)(s_agn + i);
        __syncthreads();
    }
}

extern "C" void kernel_launch(void* const* d_in, const int* in_sizes, int n_in,
                              void* d_out, int out_size, void* d_ws, size_t ws_size,
                              hipStream_t stream) {
    (void)in_sizes; (void)n_in; (void)out_size; (void)ws_size;
    const float* w_qp   = (const float*)d_in[4];
    const float* a_w_in = (const float*)d_in[12];
    const float* a_w_out= (const float*)d_in[14];
    const float* a_w1   = (const float*)d_in[16];
    const float* a_w2   = (const float*)d_in[18];
    const float* o_w_in = (const float*)d_in[26];
    const float* o_w_out= (const float*)d_in[28];
    const float* o_w1   = (const float*)d_in[30];
    const float* o_w2   = (const float*)d_in[32];

    uint4* frags = (uint4*)d_ws;
    auto pack = [&](const float* W, int N, int K, int foff) {
        const int nf = (N >> 4) * (K >> 5);
        pack_weights<<<dim3((nf + NWAVES - 1) / NWAVES), dim3(NTHREADS), 0, stream>>>(
            W, N, K, frags + (size_t)foff * 64);
    };
    pack(w_qp,                256, 512,  FR_WQP);
    pack(a_w_in,              256, 256,  FR_AWQ);
    pack(a_w_in + 1 * 65536,  256, 256,  FR_AWK);
    pack(a_w_in + 2 * 65536,  256, 256,  FR_AWV);
    pack(a_w_out,             256, 256,  FR_AWO);
    pack(a_w1,               1024, 256,  FR_AW1);
    pack(a_w2,                256, 1024, FR_AW2);
    pack(o_w_in,              256, 256,  FR_OWQ);
    pack(o_w_in + 1 * 65536,  256, 256,  FR_OWK);
    pack(o_w_in + 2 * 65536,  256, 256,  FR_OWV);
    pack(o_w_out,             256, 256,  FR_OWO);
    pack(o_w1,               1024, 256,  FR_OW1);
    pack(o_w2,                256, 1024, FR_OW2);

    Params p;
    p.hist   = (const float*)d_in[0];
    p.act    = (const float*)d_in[1];
    p.w_act  = (const float*)d_in[2];
    p.b_act  = (const float*)d_in[3];
    p.b_qp   = (const float*)d_in[5];
    p.a_qn_g = (const float*)d_in[6];
    p.a_qn_b = (const float*)d_in[7];
    p.a_kvn_g = (const float*)d_in[8];
    p.a_kvn_b = (const float*)d_in[9];
    p.a_ffn_g = (const float*)d_in[10];
    p.a_ffn_b = (const float*)d_in[11];
    p.a_b_in  = (const float*)d_in[13];
    p.a_b_out = (const float*)d_in[15];
    p.a_b1    = (const float*)d_in[17];
    p.a_b2    = (const float*)d_in[19];
    p.o_qn_g  = (const float*)d_in[20];
    p.o_qn_b  = (const float*)d_in[21];
    p.o_kvn_g = (const float*)d_in[22];
    p.o_kvn_b = (const float*)d_in[23];
    p.o_ffn_g = (const float*)d_in[24];
    p.o_ffn_b = (const float*)d_in[25];
    p.o_b_in  = (const float*)d_in[27];
    p.o_b_out = (const float*)d_in[29];
    p.o_b1    = (const float*)d_in[31];
    p.o_b2    = (const float*)d_in[33];
    p.view_emb = (const float*)d_in[34];
    p.slot_emb = (const float*)d_in[35];
    p.frags = frags;

    float* out = (float*)d_out;
    p.ps = out;
    p.po = p.ps + (size_t)Bsz * TPn * Tn * Dn;
    p.pa = p.po + (size_t)Bsz * TPn * 28 * Dn;

    hipFuncSetAttribute(reinterpret_cast<const void*>(slotpred_kernel),
                        hipFuncAttributeMaxDynamicSharedMemorySize, SMEM_BYTES);
    slotpred_kernel<<<dim3(Bsz), dim3(NTHREADS), SMEM_BYTES, stream>>>(p);
}
